// SelfAttention_44710609551425
// MI455X (gfx1250) — matrix operand order repaired and checked by reading, not compiled, not run on hardware
//
#include <hip/hip_runtime.h>
#include <hip/hip_bf16.h>

// ---------------- problem constants ----------------
#define B0      2
#define SEQ     8192
#define DIM     1024
#define HEADS   16
#define LHEADS  8
#define GHEADS  8
#define HD      64
#define BUCKET  64
#define WSZ     128
#define NBLK    (SEQ / BUCKET)   // 128
#define NCLU    (SEQ / WSZ)      // 64
#define NEGMAX  (-3.402823466e38f)
#define SELFV   (-50000.0f)
#define SCALE   (0.125f)         // 64^-0.5

typedef __attribute__((ext_vector_type(16))) __bf16 v16bf;
typedef __attribute__((ext_vector_type(8)))  float  v8f;

// ---------------- CDNA5 async Global<->LDS path (ASYNCcnt) ----------------
// Builtins take typed, address-space-qualified pointers:
//   (addrspace(1) int* gaddr, addrspace(3) int* lds, imm offset, imm cpol)
#if defined(__gfx1250__) && __has_builtin(__builtin_amdgcn_global_load_async_to_lds_b32)
#define HAVE_ASYNC_LD 1
#endif
#if defined(__gfx1250__) && __has_builtin(__builtin_amdgcn_global_store_async_from_lds_b32)
#define HAVE_ASYNC_ST 1
#endif
#if defined(HAVE_ASYNC_LD) || defined(HAVE_ASYNC_ST)
#define TO_GLOBAL_I32(p) ((__attribute__((address_space(1))) int*)(p))
#define TO_LDS_I32(p)    ((__attribute__((address_space(3))) int*)(p))
#if __has_builtin(__builtin_amdgcn_s_wait_asynccnt)
#define ASYNC_WAIT() __builtin_amdgcn_s_wait_asynccnt(0)
#else
#define ASYNC_WAIT() asm volatile("s_wait_asynccnt 0x0" ::: "memory")
#endif
#endif

// ---------------- WMMA fragment helpers (wave32; contiguous k order, same for A and B) ----------------
// A 16x32 bf16: lanes 0-15 -> M=lane, K=0..15 ; lanes 16-31 -> M=lane-16, K=16..31
//   element e : K = kbase + e
template <typename SRC>
__device__ inline v16bf fragA(const SRC* __restrict__ row, int k0) {
  int hi = (threadIdx.x >> 4) & 1;
  v16bf f;
  int kb = k0 + hi * 16;
#pragma unroll
  for (int e = 0; e < 16; ++e) f[e] = (__bf16)row[kb + e];
  return f;
}
// B 32x16 bf16 (per-column row of B^T): lanes 0-15 -> N=lane, K=0..15 ; lanes 16-31 -> K=16..31
template <typename SRC>
__device__ inline v16bf fragB(const SRC* __restrict__ row, int k0) {
  int hi = (threadIdx.x >> 4) & 1;
  v16bf f;
  int kb = k0 + hi * 16;
#pragma unroll
  for (int e = 0; e < 16; ++e) f[e] = (__bf16)row[kb + e];
  return f;
}
__device__ inline v8f wmma_bf16(v16bf a, v16bf b, v8f c) {
  return __builtin_amdgcn_wmma_f32_16x16x32_bf16(false, a, false, b, (short)0, c,
                                                 false, false);
}

// =====================================================================
// 1) Tiled bf16-WMMA GEMM:  C[M,N] = A[M,K] * W[N,K]^T (+ bias)
//    fp32 in / fp32 out, bf16 matrix-core compute. 8 waves/block,
//    each wave -> one 16(M) x 64(N) tile, K-loop step 32.
// =====================================================================
__global__ __launch_bounds__(256) void gemm_bf16_kernel(
    const float* __restrict__ A, const float* __restrict__ W,
    const float* __restrict__ bias, float* __restrict__ C, int M, int N, int K) {
  int lane = threadIdx.x & 31;
  int wv = threadIdx.x >> 5;
  int tilesN = N >> 6;
  long wid = (long)blockIdx.x * 8 + wv;
  long mt = wid / tilesN;
  int nt = (int)(wid % tilesN);
  int m0 = (int)mt * 16;
  int n0 = nt * 64;
  if (m0 >= M) return;

  const float* arow = A + (long)(m0 + (lane & 15)) * K;
  const float* wrow[4];
#pragma unroll
  for (int j = 0; j < 4; ++j) wrow[j] = W + (long)(n0 + j * 16 + (lane & 15)) * K;

  v8f acc[4] = {{}, {}, {}, {}};
  for (int k0 = 0; k0 < K; k0 += 32) {
    __builtin_prefetch(arow + k0 + 256, 0, 0);
    __builtin_prefetch(wrow[0] + k0 + 256, 0, 0);
    v16bf a = fragA(arow, k0);
#pragma unroll
    for (int j = 0; j < 4; ++j) {
      v16bf b = fragB(wrow[j], k0);
      acc[j] = wmma_bf16(a, b, acc[j]);
    }
  }
  int nn = n0 + (lane & 15);
  int mb = m0 + ((lane >> 4) << 3);
#pragma unroll
  for (int j = 0; j < 4; ++j) {
    int col = nn + j * 16;
    float bv = bias ? bias[col] : 0.f;
#pragma unroll
    for (int r = 0; r < 8; ++r) C[(long)(mb + r) * N + col] = acc[j][r] + bv;
  }
}

// =====================================================================
// 2) Local block attention (heads 0..7). One block per (b, h, nblk).
//    q: 64x64 (raw), k: 128x64 (l2norm'd window), scores 64x128 in LDS,
//    rel-shift scatter, mask+softmax, PV via WMMA.
// =====================================================================
__global__ __launch_bounds__(256) void local_attn_kernel(
    const float* __restrict__ qk, const float* __restrict__ vbuf,
    const float* __restrict__ rel, float* __restrict__ aout) {
  __shared__ __align__(16) char smem[57344];
  __bf16* qS = (__bf16*)smem;                  //  64*64 bf16  (0..8K)
  __bf16* kS = (__bf16*)(smem + 8192);         // 128*64 bf16  (8K..24K)
  float* dS = (float*)(smem + 24576);          //  64*128 f32  (24K..56K)
  __bf16* vS = (__bf16*)smem;                  // phase2: vT 64x128 bf16 (0..16K)

  int bid = blockIdx.x;
  int nb = bid % NBLK;
  int hb = bid / NBLK;
  int h = hb % LHEADS;
  int b = hb / LHEADS;
  int tid = threadIdx.x, lane = tid & 31, wv = tid >> 5;

  const float* qkh = qk + (long)b * SEQ * DIM + h * HD;
  const float* vh = vbuf + (long)b * SEQ * DIM + h * HD;

  // --- stage q (raw) and k (l2-normalized window) into LDS as bf16 ---
  if (tid < 128) {
    int r = tid;
    int tk = (nb - 1) * BUCKET + r;
    if (tk >= 0) {
      const float* src = qkh + (long)tk * DIM;
      float ss = 0.f;
      for (int d = 0; d < HD; ++d) { float x = src[d]; ss += x * x; }
      float inv = 1.f / fmaxf(sqrtf(ss), 1e-12f);
      for (int d = 0; d < HD; ++d) kS[r * HD + d] = (__bf16)(src[d] * inv);
    } else {
      for (int d = 0; d < HD; ++d) kS[r * HD + d] = (__bf16)0.f;
    }
  } else if (tid < 192) {
    int i = tid - 128;
    const float* src = qkh + (long)(nb * BUCKET + i) * DIM;
    for (int d = 0; d < HD; ++d) qS[i * HD + d] = (__bf16)src[d];
  }
  __syncthreads();

  // --- scores: 64x128 = 4 row-bands x 8 col-tiles; wave -> (rb, 4 cols) ---
  int rb = wv >> 1, cg = wv & 1;
  const __bf16* arow = qS + (rb * 16 + (lane & 15)) * HD;
  v8f acc[4] = {{}, {}, {}, {}};
  for (int k0 = 0; k0 < HD; k0 += 32) {
    v16bf a = fragA(arow, k0);
#pragma unroll
    for (int j = 0; j < 4; ++j) {
      int cb = cg * 64 + j * 16;
      v16bf bb = fragB(kS + (cb + (lane & 15)) * HD, k0);
      acc[j] = wmma_bf16(a, bb, acc[j]);
    }
  }
  {
    int mb = rb * 16 + ((lane >> 4) << 3);
#pragma unroll
    for (int j = 0; j < 4; ++j) {
      int col = cg * 64 + j * 16 + (lane & 15);
#pragma unroll
      for (int r = 0; r < 8; ++r) dS[(mb + r) * 128 + col] = acc[j][r];
    }
  }
  __syncthreads();

  // --- rel term: R[i][c] = q_i . rel[c,h,:] ; scatter dS[i][c+i-63] += R ---
  {
    v8f racc[4] = {{}, {}, {}, {}};
    for (int k0 = 0; k0 < HD; k0 += 32) {
      v16bf a = fragA(arow, k0);
#pragma unroll
      for (int j = 0; j < 4; ++j) {
        int c = cg * 64 + j * 16 + (lane & 15);
        v16bf bb = fragB(rel + ((long)c * LHEADS + h) * HD, k0);
        racc[j] = wmma_bf16(a, bb, racc[j]);
      }
    }
    int mb = rb * 16 + ((lane >> 4) << 3);
#pragma unroll
    for (int j = 0; j < 4; ++j) {
      int c = cg * 64 + j * 16 + (lane & 15);
#pragma unroll
      for (int r = 0; r < 8; ++r) {
        int m = mb + r;
        int col = c + m - 63;
        if (col >= 0 && col < 128) dS[m * 128 + col] += racc[j][r];
      }
    }
  }
  __syncthreads();

  // --- scale + masks + softmax (one thread per query row) ---
  if (tid < 64) {
    int i = tid;
    int tq = nb * BUCKET + i;
    float mx = NEGMAX;
    for (int j = 0; j < 128; ++j) {
      int tk = (nb - 1) * BUCKET + j;
      float v = dS[i * 128 + j] * SCALE;
      if (tk == tq) v = SELFV;
      else if (tk > tq || tk < 0) v = NEGMAX;
      dS[i * 128 + j] = v;
      mx = fmaxf(mx, v);
    }
    float s = 0.f;
    for (int j = 0; j < 128; ++j) {
      float e = __expf(dS[i * 128 + j] - mx);
      dS[i * 128 + j] = e;
      s += e;
    }
    float inv = 1.f / s;
    for (int j = 0; j < 128; ++j) dS[i * 128 + j] *= inv;
  }
  __syncthreads();

  // --- load v transposed (vT[e][j]) over the now-dead q/k region ---
  for (int idx = tid; idx < 128 * 64; idx += 256) {
    int j = idx >> 6, e = idx & 63;
    int tk = (nb - 1) * BUCKET + j;
    float val = (tk >= 0) ? vh[(long)tk * DIM + e] : 0.f;
    vS[e * 128 + j] = (__bf16)val;
  }
  __syncthreads();

  // --- out = attn(64x128) @ v(128x64); wave -> (mt, 2 col tiles) ---
  int mt = wv >> 1, ng = wv & 1;
  v8f oa[2] = {{}, {}};
  const float* arow2 = dS + (mt * 16 + (lane & 15)) * 128;
  for (int k0 = 0; k0 < 128; k0 += 32) {
    v16bf a = fragA(arow2, k0);
#pragma unroll
    for (int j = 0; j < 2; ++j) {
      int nbase = (ng * 2 + j) * 16;
      v16bf bb = fragB(vS + (nbase + (lane & 15)) * 128, k0);
      oa[j] = wmma_bf16(a, bb, oa[j]);
    }
  }
  float* outh = aout + (long)b * SEQ * DIM + h * HD;
  int mb2 = mt * 16 + ((lane >> 4) << 3);
#pragma unroll
  for (int j = 0; j < 2; ++j) {
    int e = (ng * 2 + j) * 16 + (lane & 15);
#pragma unroll
    for (int r = 0; r < 8; ++r) {
      int tq = nb * BUCKET + mb2 + r;
      outh[(long)tq * DIM + e] = oa[j][r];
    }
  }
}

// =====================================================================
// 3) kn = l2norm(qk global heads)  ->  kn[(b*GH+h)*SEQ + t][0..63]
// =====================================================================
__global__ __launch_bounds__(256) void l2norm_rows_kernel(
    const float* __restrict__ qk, float* __restrict__ kn) {
  long r = (long)blockIdx.x * blockDim.x + threadIdx.x;
  if (r >= (long)B0 * GHEADS * SEQ) return;
  long t = r % SEQ;
  long bh = r / SEQ;
  int h = (int)(bh % GHEADS), b = (int)(bh / GHEADS);
  const float* src = qk + ((long)b * SEQ + t) * DIM + (LHEADS + h) * HD;
  float ss = 0.f;
  for (int d = 0; d < HD; ++d) { float x = src[d]; ss += x * x; }
  float inv = 1.f / fmaxf(sqrtf(ss), 1e-12f);
  float* dst = kn + r * HD;
  for (int d = 0; d < HD; ++d) dst[d] = src[d] * inv;
}

// =====================================================================
// 4) dists[bh, t, c] = kn . means[h,c,:]   (WMMA, M=8192 N=64 K=64 per bh)
// =====================================================================
__global__ __launch_bounds__(256) void dists_kernel(
    const float* __restrict__ kn, const float* __restrict__ means,
    float* __restrict__ dists) {
  int lane = threadIdx.x & 31, wv = threadIdx.x >> 5;
  long wid = (long)blockIdx.x * 8 + wv;  // 8192 wave-tiles
  long mt = wid % (SEQ / 16);
  long bh = wid / (SEQ / 16);
  int h = (int)(bh % GHEADS);
  const float* arow = kn + (bh * SEQ + mt * 16 + (lane & 15)) * HD;
  const float* mh = means + (long)h * NCLU * HD;
  v8f acc[4] = {{}, {}, {}, {}};
  for (int k0 = 0; k0 < HD; k0 += 32) {
    v16bf a = fragA(arow, k0);
#pragma unroll
    for (int j = 0; j < 4; ++j) {
      v16bf b = fragB(mh + (long)(j * 16 + (lane & 15)) * HD, k0);
      acc[j] = wmma_bf16(a, b, acc[j]);
    }
  }
  long mbase = bh * SEQ + mt * 16 + ((lane >> 4) << 3);
#pragma unroll
  for (int j = 0; j < 4; ++j) {
    int c = j * 16 + (lane & 15);
#pragma unroll
    for (int r = 0; r < 8; ++r) dists[(mbase + r) * NCLU + c] = acc[j][r];
  }
}

// =====================================================================
// 5) top-128 per (b,h,cluster) + ascending sort of indices.
//    Column gather goes through the CDNA5 async Global->LDS engine
//    (ASYNCcnt) when available; index writeback uses async LDS->Global.
// =====================================================================
__global__ __launch_bounds__(128) void topk_kernel(const float* __restrict__ dists,
                                                   int* __restrict__ idxbuf) {
  __shared__ float vals[SEQ];       // 32 KB
  __shared__ float rmax[128];
  __shared__ int rarg[128];
  __shared__ int sel[WSZ];
  int bid = blockIdx.x;
  int c = bid % NCLU;
  long bh = bid / NCLU;
  int tid = threadIdx.x;

#ifdef HAVE_ASYNC_LD
  for (int t = tid; t < SEQ; t += 128)
    __builtin_amdgcn_global_load_async_to_lds_b32(
        TO_GLOBAL_I32(dists + (bh * SEQ + t) * NCLU + c), TO_LDS_I32(&vals[t]), 0, 0);
  ASYNC_WAIT();
#else
  for (int t = tid; t < SEQ; t += 128) vals[t] = dists[(bh * SEQ + t) * NCLU + c];
#endif
  __syncthreads();

  for (int it = 0; it < WSZ; ++it) {
    float bm = NEGMAX;
    int ba = 0;
    for (int t = tid; t < SEQ; t += 128) {
      float v = vals[t];
      if (v > bm) { bm = v; ba = t; }
    }
    rmax[tid] = bm;
    rarg[tid] = ba;
    __syncthreads();
    for (int s = 64; s > 0; s >>= 1) {
      if (tid < s && rmax[tid + s] > rmax[tid]) {
        rmax[tid] = rmax[tid + s];
        rarg[tid] = rarg[tid + s];
      }
      __syncthreads();
    }
    if (tid == 0) {
      sel[it] = rarg[0];
      vals[rarg[0]] = NEGMAX;
    }
    __syncthreads();
  }
  // odd-even transposition sort ascending
  for (int ph = 0; ph < WSZ; ++ph) {
    int i = 2 * tid + (ph & 1);
    if (tid < 64 && i + 1 < WSZ) {
      int a = sel[i], bb2 = sel[i + 1];
      if (a > bb2) { sel[i] = bb2; sel[i + 1] = a; }
    }
    __syncthreads();
  }
#ifdef HAVE_ASYNC_ST
  if (tid < WSZ)
    __builtin_amdgcn_global_store_async_from_lds_b32(
        TO_GLOBAL_I32(idxbuf + (long)bid * WSZ + tid), TO_LDS_I32(&sel[tid]), 0, 0);
  ASYNC_WAIT();
#else
  if (tid < WSZ) idxbuf[(long)bid * WSZ + tid] = sel[tid];
#endif
}

// =====================================================================
// 6) zero scratch
// =====================================================================
__global__ void zero_kernel(float* __restrict__ p, long n) {
  long i = (long)blockIdx.x * blockDim.x + threadIdx.x;
  long stride = (long)gridDim.x * blockDim.x;
  for (; i < n; i += stride) p[i] = 0.f;
}

// =====================================================================
// 7) k-means window attention. One block per (b,h,cluster,quarter):
//    32 queries x 128 keys. WMMA scores + rel-shift + softmax + WMMA PV,
//    atomic scatter into numer/denom.
// =====================================================================
__global__ __launch_bounds__(256) void kmeans_attn_kernel(
    const float* __restrict__ qk, const float* __restrict__ vbuf,
    const float* __restrict__ relg, const int* __restrict__ idxbuf,
    float* __restrict__ numer, float* __restrict__ denom) {
  __shared__ int idxS[WSZ];
  __shared__ __bf16 kS[WSZ * HD];   // 16 KB  (l2-normalized gathered keys)
  __shared__ __bf16 vS[HD * WSZ];   // 16 KB  (gathered v, transposed)
  __shared__ __bf16 qS[32 * HD];    //  4 KB  (raw gathered queries, quarter)
  __shared__ float dS[32 * WSZ];    // 16 KB  (scores)

  int bid = blockIdx.x;
  int qtr = bid & 3;
  int rest = bid >> 2;
  int cl = rest % NCLU;
  long bh = rest / NCLU;
  int h = (int)(bh % GHEADS), b = (int)(bh / GHEADS);
  int tid = threadIdx.x, lane = tid & 31, wv = tid >> 5;

  const int* ib = idxbuf + (bh * NCLU + cl) * WSZ;
#ifdef HAVE_ASYNC_LD
  if (tid < WSZ)
    __builtin_amdgcn_global_load_async_to_lds_b32(TO_GLOBAL_I32(ib + tid),
                                                  TO_LDS_I32(&idxS[tid]), 0, 0);
  ASYNC_WAIT();
#else
  if (tid < WSZ) idxS[tid] = ib[tid];
#endif
  __syncthreads();

  const float* qkh = qk + (long)b * SEQ * DIM + (LHEADS + h) * HD;
  const float* vh = vbuf + (long)b * SEQ * DIM + (LHEADS + h) * HD;

  if (tid < WSZ) {
    int r = tid;
    const float* src = qkh + (long)idxS[r] * DIM;
    float ss = 0.f;
    for (int d = 0; d < HD; ++d) { float x = src[d]; ss += x * x; }
    float inv = 1.f / fmaxf(sqrtf(ss), 1e-12f);
    for (int d = 0; d < HD; ++d) kS[r * HD + d] = (__bf16)(src[d] * inv);
  } else if (tid < WSZ + 32) {
    int i = tid - WSZ;
    const float* src = qkh + (long)idxS[qtr * 32 + i] * DIM;
    for (int d = 0; d < HD; ++d) qS[i * HD + d] = (__bf16)src[d];
  }
  for (int idx = tid; idx < WSZ * HD; idx += 256) {
    int j = idx >> 6, e = idx & 63;
    vS[e * WSZ + j] = (__bf16)vh[(long)idxS[j] * DIM + e];
  }
  __syncthreads();

  // --- scores 32x128: 2 row-bands x 8 col-tiles; wave -> (rb, 2 cols) ---
  int rb = wv >> 2, cg = wv & 3;
  const __bf16* arow = qS + (rb * 16 + (lane & 15)) * HD;
  v8f acc[2] = {{}, {}};
  for (int k0 = 0; k0 < HD; k0 += 32) {
    v16bf a = fragA(arow, k0);
#pragma unroll
    for (int j = 0; j < 2; ++j) {
      int cb = (cg * 2 + j) * 16;
      v16bf bb = fragB(kS + (cb + (lane & 15)) * HD, k0);
      acc[j] = wmma_bf16(a, bb, acc[j]);
    }
  }
  {
    int mb = rb * 16 + ((lane >> 4) << 3);
#pragma unroll
    for (int j = 0; j < 2; ++j) {
      int col = (cg * 2 + j) * 16 + (lane & 15);
#pragma unroll
      for (int r = 0; r < 8; ++r) dS[(mb + r) * WSZ + col] = acc[j][r];
    }
  }
  __syncthreads();

  // --- rel: R[i][c] = q_i . relg[c,h,:]; scatter dS[i][c + o_r - 127] += R ---
  {
    v8f racc[2] = {{}, {}};
    for (int k0 = 0; k0 < HD; k0 += 32) {
      v16bf a = fragA(arow, k0);
#pragma unroll
      for (int j = 0; j < 2; ++j) {
        int c = (cg * 2 + j) * 16 + (lane & 15);
        v16bf bb = fragB(relg + ((long)c * GHEADS + h) * HD, k0);
        racc[j] = wmma_bf16(a, bb, racc[j]);
      }
    }
    int mb = rb * 16 + ((lane >> 4) << 3);
#pragma unroll
    for (int j = 0; j < 2; ++j) {
      int c = (cg * 2 + j) * 16 + (lane & 15);
#pragma unroll
      for (int r = 0; r < 8; ++r) {
        int m = mb + r;
        int o_r = qtr * 32 + m;
        int col = c + o_r - 127;
        if (col >= 0 && col < WSZ) dS[m * WSZ + col] += racc[j][r];
      }
    }
  }
  __syncthreads();

  // --- scale + eye mask + softmax ---
  if (tid < 32) {
    int i = tid;
    int o_r = qtr * 32 + i;
    float mx = NEGMAX;
    for (int j = 0; j < WSZ; ++j) {
      float v = dS[i * WSZ + j] * SCALE;
      if (j == o_r) v = SELFV;
      dS[i * WSZ + j] = v;
      mx = fmaxf(mx, v);
    }
    float s = 0.f;
    for (int j = 0; j < WSZ; ++j) {
      float e = __expf(dS[i * WSZ + j] - mx);
      dS[i * WSZ + j] = e;
      s += e;
    }
    float inv = 1.f / s;
    for (int j = 0; j < WSZ; ++j) dS[i * WSZ + j] *= inv;
  }
  __syncthreads();

  // --- PV: 32x64 = 2 row-bands x 4 col tiles; wave -> one tile ---
  int rb2 = wv >> 2, ntile = wv & 3;
  v8f oa = {};
  const float* arow2 = dS + (rb2 * 16 + (lane & 15)) * WSZ;
  for (int k0 = 0; k0 < WSZ; k0 += 32) {
    v16bf a = fragA(arow2, k0);
    v16bf bb = fragB(vS + (ntile * 16 + (lane & 15)) * WSZ, k0);
    oa = wmma_bf16(a, bb, oa);
  }
  int e = ntile * 16 + (lane & 15);
  int mb3 = rb2 * 16 + ((lane >> 4) << 3);
#pragma unroll
  for (int r = 0; r < 8; ++r) {
    int o_r = qtr * 32 + mb3 + r;
    long tok = idxS[o_r];
    atomicAdd(&numer[(bh * SEQ + tok) * HD + e], oa[r]);
    if (ntile == 0 && (lane & 15) == 0) atomicAdd(&denom[bh * SEQ + tok], 1.0f);
  }
}

// =====================================================================
// 8) combine numer/denom into attn_out global-head columns
// =====================================================================
__global__ __launch_bounds__(256) void combine_kernel(
    const float* __restrict__ numer, const float* __restrict__ denom,
    float* __restrict__ aout) {
  long i = (long)blockIdx.x * blockDim.x + threadIdx.x;
  if (i >= (long)B0 * GHEADS * SEQ * HD) return;
  int e = (int)(i % HD);
  long r = i / HD;
  long t = r % SEQ;
  long bh = r / SEQ;
  int h = (int)(bh % GHEADS), b = (int)(bh / GHEADS);
  float val = numer[i] / (denom[r] + 1e-5f);
  aout[((long)b * SEQ + t) * DIM + (LHEADS + h) * HD + e] = val;
}

// =====================================================================
// launcher
// =====================================================================
extern "C" void kernel_launch(void* const* d_in, const int* in_sizes, int n_in,
                              void* d_out, int out_size, void* d_ws, size_t ws_size,
                              hipStream_t stream) {
  const float* x = (const float*)d_in[0];
  const float* w_qk = (const float*)d_in[1];
  const float* w_v = (const float*)d_in[2];
  const float* w_out = (const float*)d_in[3];
  const float* b_out = (const float*)d_in[4];
  const float* rel_local = (const float*)d_in[5];
  const float* rel_global = (const float*)d_in[6];
  const float* means = (const float*)d_in[7];
  float* out = (float*)d_out;

  // workspace carve-up (floats)
  float* ws = (float*)d_ws;
  const long NQK = (long)B0 * SEQ * DIM;          // 16,777,216
  const long NG = (long)B0 * GHEADS * SEQ * HD;   //  8,388,608
  float* qkb = ws;
  float* vb = qkb + NQK;
  float* aout = vb + NQK;
  float* kn = aout + NQK;
  float* dists = kn + NG;
  float* numer = dists + NG;
  float* denom = numer + NG;                       // B0*GH*SEQ = 131072
  int* idxb = (int*)(denom + (long)B0 * GHEADS * SEQ);

  const int M = B0 * SEQ;  // 16384

  // 1-2) qk / v projections
  gemm_bf16_kernel<<<2048, 256, 0, stream>>>(x, w_qk, nullptr, qkb, M, DIM, DIM);
  gemm_bf16_kernel<<<2048, 256, 0, stream>>>(x, w_v, nullptr, vb, M, DIM, DIM);

  // 3) local attention heads 0..7
  local_attn_kernel<<<B0 * LHEADS * NBLK, 256, 0, stream>>>(qkb, vb, rel_local, aout);

  // 4-6) routing: l2norm, dists, top-k
  l2norm_rows_kernel<<<(B0 * GHEADS * SEQ) / 256, 256, 0, stream>>>(qkb, kn);
  dists_kernel<<<(B0 * GHEADS * (SEQ / 16)) / 8, 256, 0, stream>>>(kn, means, dists);
  topk_kernel<<<B0 * GHEADS * NCLU, 128, 0, stream>>>(dists, idxb);

  // 7) zero accumulators (numer + denom contiguous)
  zero_kernel<<<4096, 256, 0, stream>>>(numer, NG + (long)B0 * GHEADS * SEQ);

  // 8) k-means window attention + scatter
  kmeans_attn_kernel<<<B0 * GHEADS * NCLU * 4, 256, 0, stream>>>(
      qkb, vb, rel_global, idxb, numer, denom);

  // 9) combine into attn_out global-head columns
  combine_kernel<<<(int)((NG + 255) / 256), 256, 0, stream>>>(numer, denom, aout);

  // 10) output projection + bias
  gemm_bf16_kernel<<<2048, 256, 0, stream>>>(aout, w_out, b_out, out, M, DIM, DIM);
}
